// BestChangeLayer_65532611002596
// MI455X (gfx1250) — 8-bit matrix operand order repaired and checked by reading, compiled, not run on hardware
//
#include <hip/hip_runtime.h>

typedef __attribute__((ext_vector_type(8))) int v8i;

#define NPAT 512
#define BDIM 512

__device__ __forceinline__ int mod25(int v) { v %= 25; return v < 0 ? v + 25 : v; }

__device__ __forceinline__ int adjA(int row, int j) {
  if (row >= 25) return 0;
  int r = 1 + row / 5, c = 1 + row % 5;
  int rj = 2 + j / 3, cj = 2 + j % 3;
  int dr = rj - r, dc = cj - c;
  if (dr == 0 && dc == 0) return 0;
  return (dr >= -1 && dr <= 1 && dc >= -1 && dc <= 1) ? 1 : 0;
}

__device__ __forceinline__ int pbit(int j, int p) { return (p >> (8 - j)) & 1; }

__global__ __launch_bounds__(BDIM)
void best_change_kernel(const float* __restrict__ x,
                        const float* __restrict__ target,
                        const float* __restrict__ noise,
                        const int* __restrict__ ryp,
                        const int* __restrict__ rxp,
                        float* __restrict__ out) {
  __shared__ unsigned char sint[25 * NPAT];
  __shared__ int sring[25];
  __shared__ int xcl[25];
  __shared__ int twl[25];
  __shared__ unsigned long long bestkey;

  const int tid = threadIdx.x;
  const int b = blockIdx.x;
  const int ry = ryp[0], rx = rxp[0];
  const float* xb = x + b * 625;
  const float* tb = target + b * 625;

  if (tid < 32) {
    const int lmod = tid & 15;
    const int hi = tid >> 4;
    for (int mt = 0; mt < 2; ++mt) {
      const int row = mt * 16 + lmod;
      unsigned a0, a1;
      if (hi == 0) {
        a0 = (unsigned)adjA(row, 0) | ((unsigned)adjA(row, 1) << 8) |
             ((unsigned)adjA(row, 2) << 16) | ((unsigned)adjA(row, 3) << 24);
        a1 = (unsigned)adjA(row, 4) | ((unsigned)adjA(row, 5) << 8) |
             ((unsigned)adjA(row, 6) << 16) | ((unsigned)adjA(row, 7) << 24);
      } else {
        a0 = (unsigned)adjA(row, 8);
        a1 = 0u;
      }
      v8i av = {};
      av[0] = (int)a0; av[1] = (int)a1;
      for (int t = 0; t < 32; ++t) {
        const int p = t * 16 + lmod;
        unsigned b0 = 0, b1 = 0, b2 = 0;
        if (hi == 0) {
          b0 = (unsigned)pbit(0, p) | ((unsigned)pbit(1, p) << 8) |
               ((unsigned)pbit(2, p) << 16) | ((unsigned)pbit(3, p) << 24);
          b1 = (unsigned)pbit(4, p) | ((unsigned)pbit(5, p) << 8) |
               ((unsigned)pbit(6, p) << 16) | ((unsigned)pbit(7, p) << 24);
        } else {
          b0 = (unsigned)pbit(8, p);
        }
        v8i bv = {};
        bv[0] = (int)b0; bv[1] = (int)b1; bv[2] = (int)b2;
        v8i cz = {};
        v8i d = __builtin_amdgcn_wmma_i32_16x16x64_iu8(false, av, false, bv,
                                                       cz, false, false);
#pragma unroll
        for (int g = 0; g < 8; ++g) {
          const int orow = mt * 16 + g + hi * 8;
          if (orow < 25) sint[orow * NPAT + p] = (unsigned char)d[g];
        }
      }
    }
  } else if (tid < 57) {
    const int c = tid - 32;
    const int a5 = c / 5, b5 = c % 5;
    const int u = a5 + 1, v = b5 + 1;
    int s = 0;
#pragma unroll
    for (int du = -1; du <= 1; ++du)
#pragma unroll
      for (int dv = -1; dv <= 1; ++dv) {
        if (du == 0 && dv == 0) continue;
        const int nu = u + du, nv = v + dv;
        const bool interior = (nu >= 2 && nu <= 4 && nv >= 2 && nv <= 4);
        if (!interior) {
          const int rr = mod25(ry + nu - 2), cc = mod25(rx + nv - 2);
          s += (xb[rr * 25 + cc] > 0.5f) ? 1 : 0;
        }
      }
    sring[c] = s;
    const bool cint = (u >= 2 && u <= 4 && v >= 2 && v <= 4);
    int xcv = 0;
    if (!cint) {
      const int rr = mod25(ry + u - 2), cc = mod25(rx + v - 2);
      xcv = (xb[rr * 25 + cc] > 0.5f) ? 1 : 0;
    }
    xcl[c] = xcv;
    const int rr = mod25(ry + a5 - 1), cc = mod25(rx + b5 - 1);
    twl[c] = (tb[rr * 25 + cc] > 0.5f) ? 1 : 0;
  }
  if (tid == 63) bestkey = 0xFFFFFFFFFFFFFFFFull;
  __syncthreads();

  {
    const int p = tid;
    int err = 0;
#pragma unroll
    for (int c = 0; c < 25; ++c) {
      const int a5 = c / 5, b5 = c % 5;
      const int u = a5 + 1, v = b5 + 1;
      const int s = sring[c] + (int)sint[c * NPAT + p];
      int xc;
      if (u >= 2 && u <= 4 && v >= 2 && v <= 4) {
        const int j = (u - 2) * 3 + (v - 2);
        xc = (p >> (8 - j)) & 1;
      } else {
        xc = xcl[c];
      }
      const int e = ((s + xc) >= 3 && s <= 3) ? 1 : 0;
      err += e ^ twl[c];
    }
    const float seeded =
        __fadd_rn((float)err, __fmul_rn(noise[b * NPAT + p], 0.5f));
    const unsigned long long key =
        ((unsigned long long)__float_as_uint(seeded) << 32) | (unsigned)p;
    atomicMin(&bestkey, key);
  }
  __syncthreads();
  const int bestp = (int)(unsigned)(bestkey & 0xFFFFFFFFull);

  float* ob = out + b * 625;
  for (int i = tid; i < 625; i += BDIM) {
    const int r = i / 25, c = i % 25;
    float val;
    if (r >= ry && r < ry + 3 && c >= rx && c < rx + 3) {
      const int j = (r - ry) * 3 + (c - rx);
      val = ((bestp >> (8 - j)) & 1) ? 1.0f : 0.0f;
    } else {
      val = xb[i];
    }
    ob[i] = val;
  }
}

extern "C" void kernel_launch(void* const* d_in, const int* in_sizes, int n_in,
                              void* d_out, int out_size, void* d_ws, size_t ws_size,
                              hipStream_t stream) {
  (void)n_in; (void)d_ws; (void)ws_size; (void)out_size;
  const float* x = (const float*)d_in[0];
  const float* target = (const float*)d_in[1];
  const float* noise = (const float*)d_in[2];
  const int* ry = (const int*)d_in[3];
  const int* rx = (const int*)d_in[4];
  float* out = (float*)d_out;
  const int B = in_sizes[0] / 625;
  best_change_kernel<<<B, BDIM, 0, stream>>>(x, target, noise, ry, rx, out);
}
